// CustomCrossAttnProcessor_75737453297853
// MI455X (gfx1250) — matrix operand order repaired and checked by reading, not compiled, not run on hardware
//
#include <hip/hip_runtime.h>
#include <hip/hip_bf16.h>

// ---------------------------------------------------------------------------
// Types / helpers
// ---------------------------------------------------------------------------
typedef unsigned short u16t;
typedef __attribute__((ext_vector_type(16))) __bf16 v16bf;
typedef __attribute__((ext_vector_type(8)))  float  v8f;

union FragB16 { uint4 u[2]; v16bf v; };

__device__ __forceinline__ u16t f2b(float f) {
  unsigned u = __float_as_uint(f);
  u += 0x7FFFu + ((u >> 16) & 1u);          // round-to-nearest-even
  return (u16t)(u >> 16);
}
__device__ __forceinline__ float b2f(u16t h) {
  return __uint_as_float(((unsigned)h) << 16);
}
__device__ __forceinline__ v8f vzero8() {
  v8f z;
#pragma unroll
  for (int i = 0; i < 8; ++i) z[i] = 0.0f;
  return z;
}
__device__ __forceinline__ v8f wmma_bf16(v16bf a, v16bf b, v8f c) {
  return __builtin_amdgcn_wmma_f32_16x16x32_bf16(false, a, false, b,
                                                 (short)0, c, false, false);
}

// Async global -> LDS copy (16B per lane), ASYNCcnt-tracked (cdna5 08_async_tensor)
__device__ __forceinline__ void async_b128(void* lds_ptr, const void* gptr) {
  unsigned lds_off = (unsigned)(unsigned long long)lds_ptr; // low 32b = LDS offset
  unsigned long long ga = (unsigned long long)gptr;
  asm volatile("global_load_async_to_lds_b128 %0, %1, off"
               :: "v"(lds_off), "v"(ga) : "memory");
}
__device__ __forceinline__ void wait_async0() {
  asm volatile("s_wait_asynccnt 0x0" ::: "memory");
}

#define NB        8
#define S_LEN     4096
#define T_LEN     77
#define T_PAD     80
#define KV_PAD    96
#define P_LEN     16
#define P_PAD     32
#define HD        64
#define DMODEL    1280
#define CDIM      2048
#define NHEADS    20

// ---------------------------------------------------------------------------
// f32 -> bf16 elementwise convert
// ---------------------------------------------------------------------------
__global__ void k_f32_to_bf16(const float* __restrict__ src,
                              u16t* __restrict__ dst, int n) {
  int i = blockIdx.x * blockDim.x + threadIdx.x;
  if (i < n) dst[i] = f2b(src[i]);
}

// f32 (R x C) -> bf16 transposed (C x R): weights staged as B^T (N x K)
__global__ void k_f32_to_bf16_tr(const float* __restrict__ src,
                                 u16t* __restrict__ dst, int R, int C) {
  int i = blockIdx.x * blockDim.x + threadIdx.x;
  if (i < R * C) {
    int r = i / C, c = i % C;
    dst[(size_t)c * R + r] = f2b(src[i]);
  }
}

// ---------------------------------------------------------------------------
// Tiled bf16 WMMA GEMM: C[MxN] = A[MxK] * B[KxN], B passed TRANSPOSED (NxK).
//   block tile 128(M) x 64(N), 8 waves, wave = 16x64, K step 32.
//   Double-buffered LDS, tiles staged with GLOBAL_LOAD_ASYNC_TO_LDS_B128.
// ---------------------------------------------------------------------------
__global__ __launch_bounds__(256)
void k_gemm_bf16(const u16t* __restrict__ A, const u16t* __restrict__ BT,
                 u16t* __restrict__ Cb, float* __restrict__ Cf,
                 const float* __restrict__ bias, int M, int N, int K) {
  __shared__ __align__(16) u16t Alds[2][128 * 32];  // [m][k], k contiguous
  __shared__ __align__(16) u16t Blds[2][64 * 32];   // [n][k], k contiguous

  const int tid  = threadIdx.x;
  const int lane = tid & 31;
  const int wave = tid >> 5;
  const int l15  = lane & 15;
  const int hi   = (lane & 16) ? 1 : 0;
  const int a0   = hi ? 8 : 0;        // A-frag K base (ISA 16-bit A layout)
  const int kb   = hi ? 8 : 0;        // B-frag K base (same table order as A)
  const int mBase = blockIdx.y * 128;
  const int nBase = blockIdx.x * 64;

  // per-thread staging assignment (16B chunks, contiguous in K)
  const int arow = tid >> 1;                 // 0..127
  const int acol = (tid & 1) * 16;           // 0 / 16
  const int gr   = mBase + arow;
  const bool aval = gr < M;
  const u16t* aptr = A + (size_t)gr * K + acol;
  const int bn = tid >> 2;                   // 0..63
  const int bk = (tid & 3) * 8;              // 0..24
  const u16t* bptr = BT + (size_t)(nBase + bn) * K + bk;

  // zero-fill A rows that are out of range (stays valid across all K tiles)
  if (!aval) {
    uint4 z = make_uint4(0, 0, 0, 0);
#pragma unroll
    for (int bfi = 0; bfi < 2; ++bfi) {
      *(uint4*)&Alds[bfi][arow * 32 + acol]     = z;
      *(uint4*)&Alds[bfi][arow * 32 + acol + 8] = z;
    }
  }

  auto issue = [&](int buf, int k0) {
    if (aval) {
      async_b128(&Alds[buf][arow * 32 + acol],     aptr + k0);
      async_b128(&Alds[buf][arow * 32 + acol + 8], aptr + k0 + 8);
      __builtin_prefetch(aptr + k0 + 64, 0, 1);   // global_prefetch_b8
    }
    async_b128(&Blds[buf][bn * 32 + bk], bptr + k0);
  };

  v8f acc[4];
#pragma unroll
  for (int i = 0; i < 4; ++i) acc[i] = vzero8();

  issue(0, 0);
  wait_async0();
  __syncthreads();

  int buf = 0;
  for (int k0 = 0; k0 < K; k0 += 32) {
    const int nbuf = buf ^ 1;
    if (k0 + 32 < K) issue(nbuf, k0 + 32);   // overlap next tile with WMMA

    FragB16 af;
    const int aRow = wave * 16 + l15;
    af.u[0] = *(const uint4*)&Alds[buf][aRow * 32 + a0];
    af.u[1] = *(const uint4*)&Alds[buf][aRow * 32 + a0 + 16];
#pragma unroll
    for (int nb = 0; nb < 4; ++nb) {
      const int bCol = nb * 16 + l15;
      FragB16 bf;
      bf.u[0] = *(const uint4*)&Blds[buf][bCol * 32 + kb];
      bf.u[1] = *(const uint4*)&Blds[buf][bCol * 32 + kb + 16];
      acc[nb] = wmma_bf16(af.v, bf.v, acc[nb]);
    }
    wait_async0();
    __syncthreads();
    buf = nbuf;
  }

#pragma unroll
  for (int nb = 0; nb < 4; ++nb) {
    const int n = nBase + nb * 16 + l15;
#pragma unroll
    for (int r = 0; r < 8; ++r) {
      const int m = mBase + wave * 16 + r + (hi ? 8 : 0);
      if (m < M) {
        float v = acc[nb][r];
        if (Cf) Cf[(size_t)m * N + n] = v + (bias ? bias[n] : 0.0f);
        else    Cb[(size_t)m * N + n] = f2b(v);
      }
    }
  }
}

// ---------------------------------------------------------------------------
// Text cross-attention: per (b,h), 128 S-rows per block; T=77 (pad 80/96)
// ---------------------------------------------------------------------------
__global__ __launch_bounds__(256)
void k_attn_text(const u16t* __restrict__ Q, const u16t* __restrict__ Kg,
                 const u16t* __restrict__ Vg, u16t* __restrict__ Lat) {
  __shared__ __align__(16) u16t Klds[T_PAD * HD];     // [t][d]
  __shared__ __align__(16) u16t Vt[HD * KV_PAD];      // [d][t], t padded to 96
  __shared__ __align__(16) char SP[128 * 320];        // f32 scores -> bf16 probs

  const int tid  = threadIdx.x;
  const int lane = tid & 31;
  const int wave = tid >> 5;
  const int l15  = lane & 15;
  const int hi   = (lane & 16) ? 1 : 0;
  const int a0   = hi ? 8 : 0;
  const int kb   = hi ? 8 : 0;
  const int sBase = blockIdx.x * 128;
  const int h = blockIdx.y;
  const int b = blockIdx.z;

  // K tile: async global->LDS for valid rows; zero-fill pad rows 77..79
  for (int i = tid; i < (T_PAD - T_LEN) * HD; i += 256) Klds[T_LEN * HD + i] = 0;
  for (int i = tid * 8; i < T_LEN * HD; i += 256 * 8) {
    int t = i / HD, d = i % HD;
    async_b128(&Klds[i], Kg + (size_t)(b * T_LEN + t) * DMODEL + h * HD + d);
  }
  // V transposed into [d][t] (t padded to 96)
  for (int i = tid; i < HD * KV_PAD; i += 256) {
    int d = i / KV_PAD, t = i % KV_PAD;
    u16t v = 0;
    if (t < T_LEN) v = Vg[(size_t)(b * T_LEN + t) * DMODEL + h * HD + d];
    Vt[i] = v;
  }
  wait_async0();
  __syncthreads();

  // scores = Q * K^T  (M=16/wave, N=80, K=64)
  v8f sc[5];
#pragma unroll
  for (int i = 0; i < 5; ++i) sc[i] = vzero8();
  const u16t* qp = Q + (size_t)(b * S_LEN + sBase + wave * 16 + l15) * DMODEL + h * HD;
#pragma unroll
  for (int kk = 0; kk < 2; ++kk) {
    FragB16 af;
    af.u[0] = *(const uint4*)(qp + kk * 32 + a0);
    af.u[1] = *(const uint4*)(qp + kk * 32 + a0 + 16);
#pragma unroll
    for (int nb = 0; nb < 5; ++nb) {
      int t = nb * 16 + l15;
      FragB16 bf;
      bf.u[0] = *(const uint4*)&Klds[t * HD + kk * 32 + kb];
      bf.u[1] = *(const uint4*)&Klds[t * HD + kk * 32 + kb + 16];
      sc[nb] = wmma_bf16(af.v, bf.v, sc[nb]);
    }
  }
  const float scale = 0.125f;  // 64^-0.5
#pragma unroll
  for (int nb = 0; nb < 5; ++nb) {
    int t = nb * 16 + l15;
#pragma unroll
    for (int r = 0; r < 8; ++r) {
      int row = wave * 16 + r + (hi ? 8 : 0);
      float v = sc[nb][r] * scale;
      if (t >= T_LEN) v = -3.0e38f;
      ((float*)(SP + row * 320))[t] = v;
    }
  }
  __syncthreads();

  if (tid < 128) { // row softmax, repack probs as bf16 in place (pad to 96)
    float* srow = (float*)(SP + tid * 320);
    float mx = -3.0e38f;
    for (int j = 0; j < T_PAD; ++j) mx = fmaxf(mx, srow[j]);
    float sum = 0.0f;
    for (int j = 0; j < T_PAD; ++j) { float e = __expf(srow[j] - mx); srow[j] = e; sum += e; }
    float inv = 1.0f / sum;
    u16t* prow = (u16t*)(SP + tid * 320);
    for (int j = 0; j < T_PAD; ++j)  prow[j] = f2b(srow[j] * inv); // 2j write < 4j read: safe
    for (int j = T_PAD; j < KV_PAD; ++j) prow[j] = 0;
  }
  __syncthreads();

  // out = probs * V  (K=96, N=64)
  v8f ov[4];
#pragma unroll
  for (int i = 0; i < 4; ++i) ov[i] = vzero8();
  const char* prow = SP + (wave * 16 + l15) * 320;
#pragma unroll
  for (int kk = 0; kk < 3; ++kk) {
    FragB16 af;
    af.u[0] = *(const uint4*)(prow + 2 * (kk * 32 + a0));
    af.u[1] = *(const uint4*)(prow + 2 * (kk * 32 + a0 + 16));
#pragma unroll
    for (int nb = 0; nb < 4; ++nb) {
      int d = nb * 16 + l15;
      FragB16 bf;
      bf.u[0] = *(const uint4*)&Vt[d * KV_PAD + kk * 32 + kb];
      bf.u[1] = *(const uint4*)&Vt[d * KV_PAD + kk * 32 + kb + 16];
      ov[nb] = wmma_bf16(af.v, bf.v, ov[nb]);
    }
  }
#pragma unroll
  for (int nb = 0; nb < 4; ++nb) {
#pragma unroll
    for (int r = 0; r < 8; ++r) {
      int s = sBase + wave * 16 + r + (hi ? 8 : 0);
      Lat[(size_t)(b * S_LEN + s) * DMODEL + h * HD + nb * 16 + l15] = f2b(ov[nb][r]);
    }
  }
}

// ---------------------------------------------------------------------------
// IP-adapter attention: P=16 keys (pad K-dim of probs@V to 32)
// ---------------------------------------------------------------------------
__global__ __launch_bounds__(256)
void k_attn_ip(const u16t* __restrict__ Q, const u16t* __restrict__ Kg,
               const u16t* __restrict__ Vg, u16t* __restrict__ Out) {
  __shared__ __align__(16) u16t Klds[P_LEN * HD];     // [t][d]
  __shared__ __align__(16) u16t Vt[HD * P_PAD];       // [d][t], t padded to 32
  __shared__ __align__(16) char SP[128 * 64];         // f32 scores -> bf16 probs

  const int tid  = threadIdx.x;
  const int lane = tid & 31;
  const int wave = tid >> 5;
  const int l15  = lane & 15;
  const int hi   = (lane & 16) ? 1 : 0;
  const int a0   = hi ? 8 : 0;
  const int kb   = hi ? 8 : 0;
  const int sBase = blockIdx.x * 128;
  const int h = blockIdx.y;
  const int b = blockIdx.z;

  for (int i = tid * 8; i < P_LEN * HD; i += 256 * 8) {
    int t = i / HD, d = i % HD;
    async_b128(&Klds[i], Kg + (size_t)(b * P_LEN + t) * DMODEL + h * HD + d);
  }
  for (int i = tid; i < HD * P_PAD; i += 256) {
    int d = i / P_PAD, t = i % P_PAD;
    u16t v = 0;
    if (t < P_LEN) v = Vg[(size_t)(b * P_LEN + t) * DMODEL + h * HD + d];
    Vt[i] = v;
  }
  wait_async0();
  __syncthreads();

  v8f sc = vzero8();
  const u16t* qp = Q + (size_t)(b * S_LEN + sBase + wave * 16 + l15) * DMODEL + h * HD;
#pragma unroll
  for (int kk = 0; kk < 2; ++kk) {
    FragB16 af, bf;
    af.u[0] = *(const uint4*)(qp + kk * 32 + a0);
    af.u[1] = *(const uint4*)(qp + kk * 32 + a0 + 16);
    int t = l15;
    bf.u[0] = *(const uint4*)&Klds[t * HD + kk * 32 + kb];
    bf.u[1] = *(const uint4*)&Klds[t * HD + kk * 32 + kb + 16];
    sc = wmma_bf16(af.v, bf.v, sc);
  }
  const float scale = 0.125f;
#pragma unroll
  for (int r = 0; r < 8; ++r) {
    int row = wave * 16 + r + (hi ? 8 : 0);
    ((float*)(SP + row * 64))[l15] = sc[r] * scale;
  }
  __syncthreads();

  if (tid < 128) {
    float* srow = (float*)(SP + tid * 64);
    float mx = -3.0e38f;
    for (int j = 0; j < P_LEN; ++j) mx = fmaxf(mx, srow[j]);
    float sum = 0.0f;
    for (int j = 0; j < P_LEN; ++j) { float e = __expf(srow[j] - mx); srow[j] = e; sum += e; }
    float inv = 1.0f / sum;
    u16t* prow = (u16t*)(SP + tid * 64);
    for (int j = 0; j < P_LEN; ++j) prow[j] = f2b(srow[j] * inv);
    for (int j = P_LEN; j < P_PAD; ++j) prow[j] = 0;
  }
  __syncthreads();

  v8f ov[4];
#pragma unroll
  for (int i = 0; i < 4; ++i) ov[i] = vzero8();
  const char* prow = SP + (wave * 16 + l15) * 64;
  {
    FragB16 af;
    af.u[0] = *(const uint4*)(prow + 2 * a0);
    af.u[1] = *(const uint4*)(prow + 2 * (a0 + 16));
#pragma unroll
    for (int nb = 0; nb < 4; ++nb) {
      int d = nb * 16 + l15;
      FragB16 bf;
      bf.u[0] = *(const uint4*)&Vt[d * P_PAD + kb];
      bf.u[1] = *(const uint4*)&Vt[d * P_PAD + kb + 16];
      ov[nb] = wmma_bf16(af.v, bf.v, ov[nb]);
    }
  }
#pragma unroll
  for (int nb = 0; nb < 4; ++nb) {
#pragma unroll
    for (int r = 0; r < 8; ++r) {
      int s = sBase + wave * 16 + r + (hi ? 8 : 0);
      Out[(size_t)(b * S_LEN + s) * DMODEL + h * HD + nb * 16 + l15] = f2b(ov[nb][r]);
    }
  }
}

// ---------------------------------------------------------------------------
// norm_ipa + combine: hs = lat + std_lat*(ip-mean_ip)/(std_ip+eps) + mean_lat
// ---------------------------------------------------------------------------
__global__ __launch_bounds__(256)
void k_norm_combine(const u16t* __restrict__ Lat, const u16t* __restrict__ Ipo,
                    u16t* __restrict__ Hs) {
  __shared__ float red[4][256];
  const int row = blockIdx.x;
  const int tid = threadIdx.x;
  const u16t* lr = Lat + (size_t)row * DMODEL;
  const u16t* ir = Ipo + (size_t)row * DMODEL;
  float s1 = 0, s2 = 0, s3 = 0, s4 = 0;
  for (int j = tid; j < DMODEL; j += 256) {
    float a = b2f(lr[j]), c = b2f(ir[j]);
    s1 += a; s2 += a * a; s3 += c; s4 += c * c;
  }
  red[0][tid] = s1; red[1][tid] = s2; red[2][tid] = s3; red[3][tid] = s4;
  __syncthreads();
  for (int off = 128; off > 0; off >>= 1) {
    if (tid < off)
      for (int q = 0; q < 4; ++q) red[q][tid] += red[q][tid + off];
    __syncthreads();
  }
  const float invn = 1.0f / (float)DMODEL;
  float mL = red[0][0] * invn, mI = red[2][0] * invn;
  float sdL = sqrtf(fmaxf(red[1][0] * invn - mL * mL, 0.0f));
  float sdI = sqrtf(fmaxf(red[3][0] * invn - mI * mI, 0.0f));
  float k = sdL / (sdI + 1e-7f);
  for (int j = tid; j < DMODEL; j += 256) {
    float a = b2f(lr[j]), c = b2f(ir[j]);
    Hs[(size_t)row * DMODEL + j] = f2b(a + (k * (c - mI) + mL));  // IP_SCALE = 1
  }
}

// ---------------------------------------------------------------------------
// Host side
// ---------------------------------------------------------------------------
extern "C" void kernel_launch(void* const* d_in, const int* in_sizes, int n_in,
                              void* d_out, int out_size, void* d_ws, size_t ws_size,
                              hipStream_t stream) {
  const float* hid  = (const float*)d_in[0];  // (8,4096,1280)
  const float* enc  = (const float*)d_in[1];  // (8,77,2048)
  const float* ipin = (const float*)d_in[2];  // (8,16,2048)
  const float* w_q  = (const float*)d_in[3];  // (1280,1280)
  const float* w_k  = (const float*)d_in[4];  // (2048,1280)
  const float* w_v  = (const float*)d_in[5];
  const float* w_ki = (const float*)d_in[6];
  const float* w_vi = (const float*)d_in[7];
  const float* w_o  = (const float*)d_in[8];  // (1280,1280)
  const float* b_o  = (const float*)d_in[9];  // (1280,)
  float* out = (float*)d_out;

  const int MS = NB * S_LEN;        // 32768
  const int MT = NB * T_LEN;        // 616
  const int MP = NB * P_LEN;        // 128

  char* ws = (char*)d_ws;
  size_t off = 0;
  auto alloc = [&](size_t elems) -> u16t* {
    u16t* p = (u16t*)(ws + off);
    off = (off + elems * sizeof(u16t) + 255) & ~(size_t)255;
    return p;
  };
  u16t* hB   = alloc((size_t)MS * DMODEL);
  u16t* eB   = alloc((size_t)MT * CDIM);
  u16t* iB   = alloc((size_t)MP * CDIM);
  u16t* wqT  = alloc((size_t)DMODEL * DMODEL);   // (N=1280, K=1280)
  u16t* wkT  = alloc((size_t)DMODEL * CDIM);     // (N=1280, K=2048)
  u16t* wvT  = alloc((size_t)DMODEL * CDIM);
  u16t* wkiT = alloc((size_t)DMODEL * CDIM);
  u16t* wviT = alloc((size_t)DMODEL * CDIM);
  u16t* woT  = alloc((size_t)DMODEL * DMODEL);
  u16t* Qb   = alloc((size_t)MS * DMODEL);
  u16t* Kb   = alloc((size_t)MT * DMODEL);
  u16t* Vb   = alloc((size_t)MT * DMODEL);
  u16t* Kib  = alloc((size_t)MP * DMODEL);
  u16t* Vib  = alloc((size_t)MP * DMODEL);
  u16t* LatB = alloc((size_t)MS * DMODEL);
  u16t* IpoB = alloc((size_t)MS * DMODEL);
  u16t* HsB  = alloc((size_t)MS * DMODEL);

  auto conv = [&](const float* s, u16t* d, int n) {
    k_f32_to_bf16<<<(n + 255) / 256, 256, 0, stream>>>(s, d, n);
  };
  auto convT = [&](const float* s, u16t* d, int R, int C) {
    int n = R * C;
    k_f32_to_bf16_tr<<<(n + 255) / 256, 256, 0, stream>>>(s, d, R, C);
  };
  conv(hid,  hB, MS * DMODEL);
  conv(enc,  eB, MT * CDIM);
  conv(ipin, iB, MP * CDIM);
  convT(w_q,  wqT,  DMODEL, DMODEL);
  convT(w_k,  wkT,  CDIM,   DMODEL);
  convT(w_v,  wvT,  CDIM,   DMODEL);
  convT(w_ki, wkiT, CDIM,   DMODEL);
  convT(w_vi, wviT, CDIM,   DMODEL);
  convT(w_o,  woT,  DMODEL, DMODEL);

  auto gemm_b = [&](const u16t* A, const u16t* BT, u16t* C, int M, int N, int K) {
    dim3 g(N / 64, (M + 127) / 128);
    k_gemm_bf16<<<g, 256, 0, stream>>>(A, BT, C, nullptr, nullptr, M, N, K);
  };
  gemm_b(hB, wqT,  Qb,  MS, DMODEL, DMODEL);   // Q
  gemm_b(eB, wkT,  Kb,  MT, DMODEL, CDIM);     // K
  gemm_b(eB, wvT,  Vb,  MT, DMODEL, CDIM);     // V
  gemm_b(iB, wkiT, Kib, MP, DMODEL, CDIM);     // K_ip
  gemm_b(iB, wviT, Vib, MP, DMODEL, CDIM);     // V_ip

  dim3 ag(S_LEN / 128, NHEADS, NB);
  k_attn_text<<<ag, 256, 0, stream>>>(Qb, Kb,  Vb,  LatB);
  k_attn_ip  <<<ag, 256, 0, stream>>>(Qb, Kib, Vib, IpoB);

  k_norm_combine<<<MS, 256, 0, stream>>>(LatB, IpoB, HsB);

  { // out = hs @ w_out + b_out  (f32 output)
    dim3 g(DMODEL / 64, MS / 128);
    k_gemm_bf16<<<g, 256, 0, stream>>>(HsB, woT, nullptr, out, b_o,
                                       MS, DMODEL, DMODEL);
  }
}
